// NeuroSAT_56367150792708
// MI455X (gfx1250) — matrix operand order repaired and checked by reading, not compiled, not run on hardware
//
#include <hip/hip_runtime.h>
#include <cstddef>
#include <cstdint>

// ---------------------------------------------------------------------------
// NeuroSAT message passing for MI455X (gfx1250), bf16 WMMA everywhere.
//   DIM=128, 26 rounds, 8192 literals, 16384 clauses.
// All matmuls are Out = A @ B^T (+bias, +relu): A:[M,K] bf16 row-major streams
// from global; B:[N,K] bf16 row-major is staged per-block into LDS with
// double-buffered global_load_async_to_lds_b128 (ASYNCcnt). All 8 B fragments
// of a K-step are loaded before the 8 WMMAs (enforced with a sched_barrier)
// so ds-load latency overlaps the matrix pipe via staged s_wait_dscnt.
// Workspace use: ~580 MB (adj + adjT in bf16 dominate: 2 x 256 MB).
// ---------------------------------------------------------------------------

#define DIMH 128
#define NROUND 26
#define PB 8
#define NVPC 512
#define NVAR (PB * NVPC)     // 4096
#define NLIT (2 * NVAR)      // 8192 literals
#define NCLS (PB * 2048)     // 16384 clauses

#define BROWS 128            // B-tile rows staged per block (= N chunk)
#define BSTRIDE 40           // u16 elements per LDS row (80B, bank-padded)

typedef unsigned short u16;
typedef __attribute__((ext_vector_type(16))) __bf16 v16bf;
typedef __attribute__((ext_vector_type(8)))  __bf16 v8bf;
typedef __attribute__((ext_vector_type(8)))  float  v8f;

__device__ __forceinline__ u16 f2bf(float f) {
  union { float f; unsigned u; } x; x.f = f;
  unsigned u = x.u;
  u += 0x7FFFu + ((u >> 16) & 1u);   // round-to-nearest-even
  return (u16)(u >> 16);
}
__device__ __forceinline__ float bf2f(u16 h) {
  union { unsigned u; float f; } x; x.u = ((unsigned)h) << 16; return x.f;
}
__device__ __forceinline__ float sigmf(float x) { return 1.f / (1.f + __expf(-x)); }

// ---------------------------------------------------------------------------
// Async-DMA one 128x32 bf16 B tile (rows nbase..nbase+127, K slice [k,k+32))
// into LDS. Thread tid stages row tid as 4 x b128 transfers. Tracked on
// ASYNCcnt; caller must s_wait_asynccnt + barrier before consuming.
// ---------------------------------------------------------------------------
__device__ __forceinline__ void async_stage(const u16* __restrict__ B, int ldb,
                                            int nbase, int k, u16* sm, int tid)
{
  const u16* g = B + (size_t)(nbase + tid) * ldb + k;
  unsigned l = (unsigned)(uintptr_t)(sm + tid * BSTRIDE);   // low 32b = LDS offset
#pragma unroll
  for (int q = 0; q < 4; ++q) {
    asm volatile("global_load_async_to_lds_b128 %0, %1, off"
                 :: "v"(l + q * 16), "v"(g + q * 8) : "memory");
  }
}

// One 32x16 B fragment for tile t from the staged LDS tile (two ds_load_b128).
__device__ __forceinline__ v16bf lds_frag(const u16* cur, int ml, int kbh, int t)
{
  const u16* p = cur + (t * 16 + ml) * BSTRIDE + kbh;   // 16B aligned
  v8bf b0 = *(const v8bf*)(p);
  v8bf b1 = *(const v8bf*)(p + 16);
  return __builtin_shufflevector(b0, b1,
      0, 1, 2, 3, 4, 5, 6, 7, 8, 9, 10, 11, 12, 13, 14, 15);
}

// ---------------------------------------------------------------------------
// acc += A @ B^T for one 16x128 strip per wave, B via LDS double buffer.
// A fragment (16-bit 16x32 ISA layout): lanes 0-15 hold K {0..7,16..23},
// lanes 16-31 hold K {8..15,24..31} of row M = lane%16 -> two b128 loads.
// A sched_barrier pins all 16 ds_loads before the 8 WMMAs so the fragments
// stay simultaneously live and the dscnt waits are staged, not drain-to-0.
// ---------------------------------------------------------------------------
__device__ __forceinline__ void gemm_pair(v8f acc[8],
    const u16* __restrict__ A, const u16* __restrict__ B,
    int K, int ldb, int rowA, int nbase, u16* sm0, u16* sm1)
{
  const int tid  = threadIdx.x;
  const int lane = tid & 31;
  const int ml   = lane & 15;
  const int hi   = lane >> 4;
  const int klo  = hi * 8;    // A-fragment K chunk offset for this half-wave
  const int kbh  = hi * 8;    // B-fragment K chunk offset (same as A)

  const u16* ar = A + (size_t)rowA * K;

  async_stage(B, ldb, nbase, 0, sm0, tid);
  for (int k = 0, i = 0; k < K; k += 32, ++i) {
    u16* cur = (i & 1) ? sm1 : sm0;
    u16* nxt = (i & 1) ? sm0 : sm1;

    asm volatile("s_wait_asynccnt 0x0" ::: "memory");  // my DMA done
    __syncthreads();                                   // tile visible to all waves
    if (k + 32 < K) async_stage(B, ldb, nbase, k + 32, nxt, tid);

    v8bf a0 = *(const v8bf*)(ar + k + klo);
    v8bf a1 = *(const v8bf*)(ar + k + 16 + klo);
    v16bf a = __builtin_shufflevector(a0, a1,
        0, 1, 2, 3, 4, 5, 6, 7, 8, 9, 10, 11, 12, 13, 14, 15);
    if (k + 512 < K)  // stream-prefetch the adjacency row (global_prefetch_b8)
      __builtin_prefetch((const void*)(ar + k + 512), 0, 1);

    // Load the whole tile's fragments first; the scheduling fence prevents
    // the machine scheduler from sinking loads down to their WMMA uses.
    v16bf bf[8];
#pragma unroll
    for (int t = 0; t < 8; ++t) bf[t] = lds_frag(cur, ml, kbh, t);
    __builtin_amdgcn_sched_barrier(0);
#pragma unroll
    for (int t = 0; t < 8; ++t)
      acc[t] = __builtin_amdgcn_wmma_f32_16x16x32_bf16(
                   false, a, false, bf[t], (short)0, acc[t], false, false);
  }
  __syncthreads();   // buffers reusable by the next A/B pair
}

__global__ __launch_bounds__(128)
void gemm_bt_kernel(
    const u16* __restrict__ A0, const u16* __restrict__ B0, int K0, int ldb0,
    const u16* __restrict__ A1, const u16* __restrict__ B1, int K1, int ldb1, int flipMask,
    const u16* __restrict__ A2, const u16* __restrict__ B2, int K2, int ldb2,
    const float* __restrict__ bias0, const float* __restrict__ bias1,
    float* __restrict__ outF, u16* __restrict__ outB, u16* __restrict__ outT,
    int M, int N, int relu)
{
  __shared__ u16 smem[2][BROWS * BSTRIDE];   // 2 x 10 KB B-tile double buffer

  const int lane  = threadIdx.x & 31;
  const int wave  = threadIdx.x >> 5;
  const int mbase = (blockIdx.x * 4 + wave) * 16;
  const int nbase = blockIdx.y * 128;
  const int ml = lane & 15;
  const int hi = lane >> 4;

  v8f acc[8];
  const v8f vz = {0.f, 0.f, 0.f, 0.f, 0.f, 0.f, 0.f, 0.f};
#pragma unroll
  for (int t = 0; t < 8; ++t) acc[t] = vz;

  gemm_pair(acc, A0, B0, K0, ldb0, mbase + ml, nbase, smem[0], smem[1]);
  if (A1) gemm_pair(acc, A1, B1, K1, ldb1, (mbase + ml) ^ flipMask, nbase, smem[0], smem[1]);
  if (A2) gemm_pair(acc, A2, B2, K2, ldb2, mbase + ml, nbase, smem[0], smem[1]);

  // Epilogue. C/D layout: vgpr g -> (M = g + 8*hi, N = lane%16) within tile.
#pragma unroll
  for (int t = 0; t < 8; ++t) {
    int n = nbase + t * 16 + ml;
    float bias = 0.f;
    if (bias0) bias += bias0[n];
    if (bias1) bias += bias1[n];
#pragma unroll
    for (int g = 0; g < 8; ++g) {
      int m = mbase + g + hi * 8;
      float v = acc[t][g] + bias;
      if (relu) v = fmaxf(v, 0.f);
      if (outF) outF[(size_t)m * N + n] = v;
      if (outB) outB[(size_t)m * N + n] = f2bf(v);
      if (outT) outT[(size_t)n * M + m] = f2bf(v);  // contiguous 8 bf16 / lane
    }
  }
}

// ---------------------------------------------------------------------------
// adj (fp32 [NLIT, NCLS]) -> adjB bf16 same layout + adjTB bf16 transposed.
// ---------------------------------------------------------------------------
__global__ void adj_cvt_kernel(const float* __restrict__ adj,
                               u16* __restrict__ adjB, u16* __restrict__ adjTB)
{
  __shared__ float tile[32][33];
  const int c0 = blockIdx.x * 32;
  const int l0 = blockIdx.y * 32;
  const int tx = threadIdx.x, ty = threadIdx.y;   // block (32, 8)
#pragma unroll
  for (int i = 0; i < 32; i += 8) {
    int l = l0 + ty + i, c = c0 + tx;
    float v = adj[(size_t)l * NCLS + c];
    tile[ty + i][tx] = v;
    adjB[(size_t)l * NCLS + c] = f2bf(v);
  }
  __syncthreads();
#pragma unroll
  for (int i = 0; i < 32; i += 8) {
    int c = c0 + ty + i, l = l0 + tx;
    adjTB[(size_t)c * NLIT + l] = f2bf(tile[tx][ty + i]);
  }
}

__global__ void cvt_bf16_kernel(const float* __restrict__ s, u16* __restrict__ d, int n)
{
  int i = blockIdx.x * blockDim.x + threadIdx.x;
  if (i < n) d[i] = f2bf(s[i]);
}

// h = broadcast(W[:,0] + b) in bf16, c = 0 fp32
__global__ void init_state_kernel(const float* __restrict__ w, const float* __restrict__ b,
                                  u16* __restrict__ hB, float* __restrict__ c, int total)
{
  int i = blockIdx.x * blockDim.x + threadIdx.x;
  if (i >= total) return;
  int d = i & (DIMH - 1);
  hB[i] = f2bf(w[d] + b[d]);
  c[i]  = 0.f;
}

// gates [M, 4*DIM] fp32 (torch order i,f,g,o); c fp32 in/out; h out bf16
__global__ void lstm_kernel(const float* __restrict__ gates, float* __restrict__ c,
                            u16* __restrict__ hB, int total)
{
  int i = blockIdx.x * blockDim.x + threadIdx.x;
  if (i >= total) return;
  int m = i >> 7, d = i & (DIMH - 1);
  const float* g = gates + ((size_t)m << 9);
  float gi = sigmf(g[d]);
  float gf = sigmf(g[DIMH + d]);
  float gg = tanhf(g[2 * DIMH + d]);
  float go = sigmf(g[3 * DIMH + d]);
  float cn = gf * c[i] + gi * gg;
  c[i]  = cn;
  hB[i] = f2bf(go * tanhf(cn));
}

// vote: per problem p, mean over 2*NVPC rows of dot(v2[row], w3)+b3
__global__ void vote_kernel(const u16* __restrict__ v2, const float* __restrict__ w3,
                            const float* __restrict__ b3, float* __restrict__ out)
{
  __shared__ float red[256];
  const int p = blockIdx.x, t = threadIdx.x;
  float s = 0.f;
  for (int r = t; r < 2 * NVPC; r += 256) {
    int row = (r < NVPC) ? (p * NVPC + r) : (NVAR + p * NVPC + (r - NVPC));
    const u16* x = v2 + (size_t)row * DIMH;
    float d = 0.f;
    for (int k = 0; k < DIMH; ++k) d += bf2f(x[k]) * w3[k];
    s += d + b3[0];
  }
  red[t] = s;
  __syncthreads();
  for (int o = 128; o > 0; o >>= 1) {
    if (t < o) red[t] += red[t + o];
    __syncthreads();
  }
  if (t == 0) out[p] = red[0] * (1.f / (2.f * NVPC));
}

// ---------------------------------------------------------------------------
static inline void gemm(hipStream_t st, int M, int N,
    const u16* A0, const u16* B0, int K0, int ldb0,
    const u16* A1, const u16* B1, int K1, int ldb1, int flipMask,
    const u16* A2, const u16* B2, int K2, int ldb2,
    const float* b0, const float* b1, int relu,
    float* oF, u16* oB, u16* oT)
{
  dim3 grid(M / 64, N / 128);   // 4 waves/block, 16 rows/wave, 128 cols/block
  gemm_bt_kernel<<<grid, dim3(128), 0, st>>>(
      A0, B0, K0, ldb0, A1, B1, K1, ldb1, flipMask,
      A2, B2, K2, ldb2, b0, b1, oF, oB, oT, M, N, relu);
}

extern "C" void kernel_launch(void* const* d_in, const int* in_sizes, int n_in,
                              void* d_out, int out_size, void* d_ws, size_t ws_size,
                              hipStream_t stream)
{
  (void)in_sizes; (void)n_in; (void)out_size; (void)ws_size;

  const float* adj    = (const float*)d_in[0];
  // d_in[1] (n_vars_per) unused: all problems are equal-sized (NVPC).
  const float* Li_w   = (const float*)d_in[2];
  const float* Li_b   = (const float*)d_in[3];
  const float* Ci_w   = (const float*)d_in[4];
  const float* Ci_b   = (const float*)d_in[5];
  const float* Lm_w1  = (const float*)d_in[6];   const float* Lm_b1 = (const float*)d_in[7];
  const float* Lm_w2  = (const float*)d_in[8];   const float* Lm_b2 = (const float*)d_in[9];
  const float* Lm_w3  = (const float*)d_in[10];  const float* Lm_b3 = (const float*)d_in[11];
  const float* Cm_w1  = (const float*)d_in[12];  const float* Cm_b1 = (const float*)d_in[13];
  const float* Cm_w2  = (const float*)d_in[14];  const float* Cm_b2 = (const float*)d_in[15];
  const float* Cm_w3  = (const float*)d_in[16];  const float* Cm_b3 = (const float*)d_in[17];
  const float* Lv_w1  = (const float*)d_in[18];  const float* Lv_b1 = (const float*)d_in[19];
  const float* Lv_w2  = (const float*)d_in[20];  const float* Lv_b2 = (const float*)d_in[21];
  const float* Lv_w3  = (const float*)d_in[22];  const float* Lv_b3 = (const float*)d_in[23];
  const float* Lu_wih = (const float*)d_in[24];  const float* Lu_whh = (const float*)d_in[25];
  const float* Lu_bih = (const float*)d_in[26];  const float* Lu_bhh = (const float*)d_in[27];
  const float* Cu_wih = (const float*)d_in[28];  const float* Cu_whh = (const float*)d_in[29];
  const float* Cu_bih = (const float*)d_in[30];  const float* Cu_bhh = (const float*)d_in[31];

  char* base = (char*)d_ws;
  size_t off = 0;
  auto alloc = [&](size_t bytes) -> char* {
    char* p = base + off;
    off += (bytes + 255) & ~(size_t)255;   // keep 256B alignment for b128 paths
    return p;
  };

  u16*   adjTB = (u16*)  alloc((size_t)NCLS * NLIT * 2);   // 256 MB
  u16*   adjB  = (u16*)  alloc((size_t)NLIT * NCLS * 2);   // 256 MB
  u16*   LhB   = (u16*)  alloc((size_t)NLIT * DIMH * 2);
  u16*   ChB   = (u16*)  alloc((size_t)NCLS * DIMH * 2);
  float* Lc    = (float*)alloc((size_t)NLIT * DIMH * 4);
  float* Cc    = (float*)alloc((size_t)NCLS * DIMH * 4);
  u16*   t1    = (u16*)  alloc((size_t)NCLS * DIMH * 2);
  u16*   t2    = (u16*)  alloc((size_t)NCLS * DIMH * 2);
  u16*   LpreT = (u16*)  alloc((size_t)DIMH * NLIT * 2);   // [128, 8192]
  u16*   CpreT = (u16*)  alloc((size_t)DIMH * NCLS * 2);   // [128, 16384]
  u16*   LCmsg = (u16*)  alloc((size_t)NCLS * DIMH * 2);
  u16*   CLmsg = (u16*)  alloc((size_t)NLIT * DIMH * 2);
  float* gates = (float*)alloc((size_t)NCLS * 4 * DIMH * 4);  // 32 MB, shared L/C
  u16* wLm1 = (u16*)alloc(DIMH * DIMH * 2);
  u16* wLm2 = (u16*)alloc(DIMH * DIMH * 2);
  u16* wLm3 = (u16*)alloc(DIMH * DIMH * 2);
  u16* wCm1 = (u16*)alloc(DIMH * DIMH * 2);
  u16* wCm2 = (u16*)alloc(DIMH * DIMH * 2);
  u16* wCm3 = (u16*)alloc(DIMH * DIMH * 2);
  u16* wLv1 = (u16*)alloc(DIMH * DIMH * 2);
  u16* wLv2 = (u16*)alloc(DIMH * DIMH * 2);
  u16* wLuIH = (u16*)alloc((size_t)4 * DIMH * 2 * DIMH * 2);  // [512,256]
  u16* wLuHH = (u16*)alloc((size_t)4 * DIMH * DIMH * 2);      // [512,128]
  u16* wCuIH = (u16*)alloc((size_t)4 * DIMH * DIMH * 2);
  u16* wCuHH = (u16*)alloc((size_t)4 * DIMH * DIMH * 2);

  auto cvt = [&](const float* s, u16* d, int n) {
    cvt_bf16_kernel<<<(n + 255) / 256, 256, 0, stream>>>(s, d, n);
  };

  // One-time (per call) conversions: adjacency to bf16 (+ transpose), weights.
  adj_cvt_kernel<<<dim3(NCLS / 32, NLIT / 32), dim3(32, 8), 0, stream>>>(adj, adjB, adjTB);
  cvt(Lm_w1, wLm1, DIMH * DIMH);  cvt(Lm_w2, wLm2, DIMH * DIMH);  cvt(Lm_w3, wLm3, DIMH * DIMH);
  cvt(Cm_w1, wCm1, DIMH * DIMH);  cvt(Cm_w2, wCm2, DIMH * DIMH);  cvt(Cm_w3, wCm3, DIMH * DIMH);
  cvt(Lv_w1, wLv1, DIMH * DIMH);  cvt(Lv_w2, wLv2, DIMH * DIMH);
  cvt(Lu_wih, wLuIH, 4 * DIMH * 2 * DIMH);
  cvt(Lu_whh, wLuHH, 4 * DIMH * DIMH);
  cvt(Cu_wih, wCuIH, 4 * DIMH * DIMH);
  cvt(Cu_whh, wCuHH, 4 * DIMH * DIMH);

  init_state_kernel<<<NLIT * DIMH / 256, 256, 0, stream>>>(Li_w, Li_b, LhB, Lc, NLIT * DIMH);
  init_state_kernel<<<NCLS * DIMH / 256, 256, 0, stream>>>(Ci_w, Ci_b, ChB, Cc, NCLS * DIMH);

  const u16* NB = nullptr;  // null bf16 ptr

  for (int r = 0; r < NROUND; ++r) {
    // ---- L_msg MLP: L_pre = W3(relu(W2(relu(W1 Lh)))) ; last layer stored T
    gemm(stream, NLIT, DIMH, LhB, wLm1, DIMH, DIMH, NB, NB, 0, 0, 0, NB, NB, 0, 0,
         Lm_b1, nullptr, 1, nullptr, t1, nullptr);
    gemm(stream, NLIT, DIMH, t1, wLm2, DIMH, DIMH, NB, NB, 0, 0, 0, NB, NB, 0, 0,
         Lm_b2, nullptr, 1, nullptr, t2, nullptr);
    gemm(stream, NLIT, DIMH, t2, wLm3, DIMH, DIMH, NB, NB, 0, 0, 0, NB, NB, 0, 0,
         Lm_b3, nullptr, 0, nullptr, nullptr, LpreT);
    // ---- LC_msg = adj^T @ L_pre   (A streams HBM, B = LpreT via LDS DMA)
    gemm(stream, NCLS, DIMH, adjTB, LpreT, NLIT, NLIT, NB, NB, 0, 0, 0, NB, NB, 0, 0,
         nullptr, nullptr, 0, nullptr, LCmsg, nullptr);
    // ---- C LSTM gates = LC_msg@Wih^T + Ch@Whh^T + bih + bhh
    gemm(stream, NCLS, 4 * DIMH, LCmsg, wCuIH, DIMH, DIMH, ChB, wCuHH, DIMH, DIMH, 0,
         NB, NB, 0, 0, Cu_bih, Cu_bhh, 0, gates, nullptr, nullptr);
    lstm_kernel<<<NCLS * DIMH / 256, 256, 0, stream>>>(gates, Cc, ChB, NCLS * DIMH);
    // ---- C_msg MLP; last layer stored transposed
    gemm(stream, NCLS, DIMH, ChB, wCm1, DIMH, DIMH, NB, NB, 0, 0, 0, NB, NB, 0, 0,
         Cm_b1, nullptr, 1, nullptr, t1, nullptr);
    gemm(stream, NCLS, DIMH, t1, wCm2, DIMH, DIMH, NB, NB, 0, 0, 0, NB, NB, 0, 0,
         Cm_b2, nullptr, 1, nullptr, t2, nullptr);
    gemm(stream, NCLS, DIMH, t2, wCm3, DIMH, DIMH, NB, NB, 0, 0, 0, NB, NB, 0, 0,
         Cm_b3, nullptr, 0, nullptr, nullptr, CpreT);
    // ---- CL_msg = adj @ C_pre
    gemm(stream, NLIT, DIMH, adjB, CpreT, NCLS, NCLS, NB, NB, 0, 0, 0, NB, NB, 0, 0,
         nullptr, nullptr, 0, nullptr, CLmsg, nullptr);
    // ---- L LSTM gates = CL_msg@Wih[:, :128]^T + flip(Lh)@Wih[:, 128:]^T + Lh@Whh^T
    //      flip(row) = row XOR NVAR  (swaps literal polarity halves)
    gemm(stream, NLIT, 4 * DIMH,
         CLmsg, wLuIH,        DIMH, 2 * DIMH,
         LhB,   wLuIH + DIMH, DIMH, 2 * DIMH, NVAR,
         LhB,   wLuHH,        DIMH, DIMH,
         Lu_bih, Lu_bhh, 0, gates, nullptr, nullptr);
    lstm_kernel<<<NLIT * DIMH / 256, 256, 0, stream>>>(gates, Lc, LhB, NLIT * DIMH);
  }

  // ---- vote head: two relu layers via WMMA, final [1,128] dot + per-problem mean
  gemm(stream, NLIT, DIMH, LhB, wLv1, DIMH, DIMH, NB, NB, 0, 0, 0, NB, NB, 0, 0,
       Lv_b1, nullptr, 1, nullptr, t1, nullptr);
  gemm(stream, NLIT, DIMH, t1, wLv2, DIMH, DIMH, NB, NB, 0, 0, 0, NB, NB, 0, 0,
       Lv_b2, nullptr, 1, nullptr, t2, nullptr);
  vote_kernel<<<PB, 256, 0, stream>>>(t2, Lv_w3, Lv_b3, (float*)d_out);
}
